// MaskAttention_20572893347881
// MI455X (gfx1250) — matrix operand order repaired and checked by reading, not compiled, not run on hardware
//
#include <hip/hip_runtime.h>
#include <math.h>

typedef __attribute__((ext_vector_type(16))) _Float16 v16h;
typedef __attribute__((ext_vector_type(8)))  float    v8f;
typedef __attribute__((ext_vector_type(4)))  _Float16 h4;

#define B_ 4
#define L_ 2048
#define H_ 8
#define E_ 64
#define D_ 64

#define QTILE 64      // query rows per block (4 waves x 16)
#define STILE 32      // key rows per inner iteration
#define NTILE (L_ / STILE)
#define QPAD  72      // f16 row stride for Q tile (36 dwords -> conflict-free)
#define KPAD  68      // f32 row stride for K tiles (bank = 4*row+col -> conflict-free)
#define VPAD  40      // f16 row stride for V^T tile (20 dwords -> conflict-free, 16B aligned)
#define PPAD  40      // f16 row stride for P scratch

// one ds_swizzle_b32 with immediate XOR pattern instead of HIP's 4-op __shfl_xor
template <int PAT>
__device__ __forceinline__ float swz(float x) {
    return __builtin_bit_cast(float,
        __builtin_amdgcn_ds_swizzle(__builtin_bit_cast(int, x), PAT));
}
#define XOR1 0x041F   // group-of-32: xor=1, and=0x1f
#define XOR2 0x081F
#define XOR4 0x101F
#define XOR8 0x201F

__global__ __launch_bounds__(128)
void MaskAttention_wmma_kernel(const float* __restrict__ Q,
                               const float* __restrict__ Kp,
                               const float* __restrict__ Vp,
                               const unsigned char* __restrict__ Msk,
                               float* __restrict__ O)
{
    __shared__ _Float16 qs[QTILE][QPAD];
    __shared__ float    kf[2][STILE][KPAD];       // K tiles: async DMA, f32
    __shared__ _Float16 vt[2][D_][VPAD];          // V^T tiles: [d][s], f16
    __shared__ unsigned char mb[2][QTILE][STILE]; // mask tiles: async DMA
    __shared__ _Float16 ps[4][16][PPAD];

    const int tid  = threadIdx.x;
    const int wave = tid >> 5;
    const int lane = tid & 31;
    const int lh   = lane >> 4;     // lane half (0/1)
    const int ln   = lane & 15;

    const int qblk  = blockIdx.x;   // 0..31
    const int h     = blockIdx.y;   // 0..7
    const int b     = blockIdx.z;   // 0..3
    const int qbase = qblk * QTILE;

    // ---------- async stager: K f32 tile + mask bytes ----------
    auto issue_tile = [&](int sbase, int buf) {
        #pragma unroll
        for (int it = 0; it < 4; ++it) {
            const int i   = tid + 128 * it;        // 0..511 (32 rows x 16 float4)
            const int row = i >> 4;
            const int c4  = i & 15;
            const size_t gofs = (((size_t)b * L_ + sbase + row) * H_ + h) * E_ + c4 * 4;
            const unsigned lk = (unsigned)(size_t)&kf[buf][row][c4 * 4];
            asm volatile("global_load_async_to_lds_b128 %0, %1, off"
                         :: "v"(lk), "v"((unsigned long long)(size_t)(Kp + gofs)) : "memory");
        }
        {   // mask tile: 64 rows x 32 bytes = 128 x 16B
            const int row = tid >> 1, part = tid & 1;
            const unsigned lm = (unsigned)(size_t)&mb[buf][row][part * 16];
            const unsigned char* g = Msk + (size_t)(qbase + row) * L_ + sbase + part * 16;
            asm volatile("global_load_async_to_lds_b128 %0, %1, off"
                         :: "v"(lm), "v"((unsigned long long)(size_t)g) : "memory");
        }
    };

    // ---------- V tile: global->regs (prefetch), regs->LDS transposed f16 ----------
    float4 vreg[4];
    auto load_v_regs = [&](int sbase) {
        #pragma unroll
        for (int it = 0; it < 4; ++it) {
            const int i = tid + 128 * it;
            const int row = i >> 4, c4 = i & 15;
            vreg[it] = ((const float4*)(Vp + (((size_t)b * L_ + sbase + row) * H_ + h) * E_))[c4];
        }
    };
    auto store_v_lds = [&](int buf) {
        #pragma unroll
        for (int it = 0; it < 4; ++it) {
            const int i = tid + 128 * it;
            const int row = i >> 4, c4 = i & 15;   // row = s, 4*c4.. = d
            vt[buf][4 * c4 + 0][row] = (_Float16)vreg[it].x;
            vt[buf][4 * c4 + 1][row] = (_Float16)vreg[it].y;
            vt[buf][4 * c4 + 2][row] = (_Float16)vreg[it].z;
            vt[buf][4 * c4 + 3][row] = (_Float16)vreg[it].w;
        }
    };

    load_v_regs(0);
    issue_tile(0, 0);     // async K + mask for first s-tile

    // ---------- stage Q tile (f32 -> f16), synchronous, once ----------
    for (int i = tid; i < 1024; i += 128) {
        int row = i >> 4;
        int e4  = i & 15;
        const float4 v = ((const float4*)(Q + (((size_t)b * L_ + qbase + row) * H_ + h) * E_))[e4];
        h4 hv; hv.x = (_Float16)v.x; hv.y = (_Float16)v.y; hv.z = (_Float16)v.z; hv.w = (_Float16)v.w;
        *(h4*)&qs[row][e4 * 4] = hv;
    }
    store_v_lds(0);
    __syncthreads();

    // ---------- gather Q as two A fragments (E chunks 0..31 / 32..63) ----------
    v16h aq0, aq1;
    {
        const int row = wave * 16 + ln;      // A layout: M = lane % 16
        #pragma unroll
        for (int j = 0; j < 8; ++j) {
            int kk = 16 * lh + 2 * j;
            aq0[2*j]   = qs[row][kk];
            aq0[2*j+1] = qs[row][kk + 1];
            aq1[2*j]   = qs[row][32 + kk];
            aq1[2*j+1] = qs[row][32 + kk + 1];
        }
    }

    v8f acc0 = {}, acc1 = {}, acc2 = {}, acc3 = {};
    float mrow[8], lrow[8];
    #pragma unroll
    for (int r = 0; r < 8; ++r) { mrow[r] = -1e30f; lrow[r] = 0.0f; }

    const float scale = 0.125f;   // 1/sqrt(64)

    for (int t = 0; t < NTILE; ++t) {
        const int buf = t & 1;

        // wait for this wave's async fills, then rendezvous so all data is visible
        asm volatile("s_wait_asynccnt 0" ::: "memory");
        __syncthreads();

        // prefetch next tile: async K/mask into other buffer, V into registers
        if (t + 1 < NTILE) {
            issue_tile((t + 1) * STILE, buf ^ 1);
            load_v_regs((t + 1) * STILE);
        }

        // ---- gather K as B fragments (f32 LDS -> f16), contraction contiguous ----
        v16h bk00, bk01, bk10, bk11;
        {
            const float* s00 = &kf[buf][ln]     [16 * lh];
            const float* s01 = &kf[buf][ln]     [32 + 16 * lh];
            const float* s10 = &kf[buf][16 + ln][16 * lh];
            const float* s11 = &kf[buf][16 + ln][32 + 16 * lh];
            #pragma unroll
            for (int j = 0; j < 16; ++j) {
                bk00[j] = (_Float16)s00[j];
                bk01[j] = (_Float16)s01[j];
                bk10[j] = (_Float16)s10[j];
                bk11[j] = (_Float16)s11[j];
            }
        }

        v8f s0 = {}, s1 = {};
        s0 = __builtin_amdgcn_wmma_f32_16x16x32_f16(false, aq0, false, bk00, (short)0, s0, false, false);
        s0 = __builtin_amdgcn_wmma_f32_16x16x32_f16(false, aq1, false, bk01, (short)0, s0, false, false);
        s1 = __builtin_amdgcn_wmma_f32_16x16x32_f16(false, aq0, false, bk10, (short)0, s1, false, false);
        s1 = __builtin_amdgcn_wmma_f32_16x16x32_f16(false, aq1, false, bk11, (short)0, s1, false, false);

        // ---- mask + scale + online softmax (C layout: row = r + 8*lh) ----
        #pragma unroll
        for (int r = 0; r < 8; ++r) {
            const int qloc = wave * 16 + r + 8 * lh;
            float x0 = s0[r] * scale;
            float x1 = s1[r] * scale;
            if (mb[buf][qloc][ln])      x0 = -1e30f;
            if (mb[buf][qloc][16 + ln]) x1 = -1e30f;

            float mx = fmaxf(x0, x1);
            mx = fmaxf(mx, swz<XOR1>(mx));
            mx = fmaxf(mx, swz<XOR2>(mx));
            mx = fmaxf(mx, swz<XOR4>(mx));
            mx = fmaxf(mx, swz<XOR8>(mx));
            const float mn = fmaxf(mrow[r], mx);

            const float e0 = __expf(x0 - mn);
            const float e1 = __expf(x1 - mn);
            float sum = e0 + e1;
            sum += swz<XOR1>(sum);
            sum += swz<XOR2>(sum);
            sum += swz<XOR4>(sum);
            sum += swz<XOR8>(sum);

            const float corr = __expf(mrow[r] - mn);
            lrow[r] = lrow[r] * corr + sum;
            mrow[r] = mn;
            acc0[r] *= corr; acc1[r] *= corr; acc2[r] *= corr; acc3[r] *= corr;

            ps[wave][r + 8 * lh][ln]      = (_Float16)e0;
            ps[wave][r + 8 * lh][16 + ln] = (_Float16)e1;
        }

        // drain next V tile into the other LDS buffer before the rendezvous
        if (t + 1 < NTILE) store_v_lds(buf ^ 1);

        __syncthreads();   // make P (and next V^T) visible

        // ---- gather P (C layout -> A fragment, contraction = key index 0..31) ----
        v16h ap;
        #pragma unroll
        for (int j = 0; j < 8; ++j) {
            int kk = 16 * lh + 2 * j;
            ap[2*j]   = ps[wave][ln][kk];
            ap[2*j+1] = ps[wave][ln][kk + 1];
        }
        // ---- gather V^T as B fragments: 16 contiguous halves per fragment ----
        v16h bv0, bv1, bv2, bv3;
        {
            const _Float16* p0 = &vt[buf][ln]     [16 * lh];
            const _Float16* p1 = &vt[buf][16 + ln][16 * lh];
            const _Float16* p2 = &vt[buf][32 + ln][16 * lh];
            const _Float16* p3 = &vt[buf][48 + ln][16 * lh];
            #pragma unroll
            for (int j = 0; j < 16; ++j) {
                bv0[j] = p0[j]; bv1[j] = p1[j]; bv2[j] = p2[j]; bv3[j] = p3[j];
            }
        }
        acc0 = __builtin_amdgcn_wmma_f32_16x16x32_f16(false, ap, false, bv0, (short)0, acc0, false, false);
        acc1 = __builtin_amdgcn_wmma_f32_16x16x32_f16(false, ap, false, bv1, (short)0, acc1, false, false);
        acc2 = __builtin_amdgcn_wmma_f32_16x16x32_f16(false, ap, false, bv2, (short)0, acc2, false, false);
        acc3 = __builtin_amdgcn_wmma_f32_16x16x32_f16(false, ap, false, bv3, (short)0, acc3, false, false);
    }

    // ---- normalize and store: out[b, q, h, d] ----
    #pragma unroll
    for (int r = 0; r < 8; ++r) {
        const int q = qbase + wave * 16 + r + 8 * lh;
        const float invl = 1.0f / lrow[r];
        const size_t o = (((size_t)b * L_ + q) * H_ + h) * (size_t)D_;
        O[o + ln]      = acc0[r] * invl;
        O[o + 16 + ln] = acc1[r] * invl;
        O[o + 32 + ln] = acc2[r] * invl;
        O[o + 48 + ln] = acc3[r] * invl;
    }
}

extern "C" void kernel_launch(void* const* d_in, const int* in_sizes, int n_in,
                              void* d_out, int out_size, void* d_ws, size_t ws_size,
                              hipStream_t stream) {
    const float*         Q   = (const float*)d_in[0];
    const float*         K   = (const float*)d_in[1];
    const float*         V   = (const float*)d_in[2];
    const unsigned char* Msk = (const unsigned char*)d_in[3];  // jnp bool -> 1 byte/elem
    float*               O   = (float*)d_out;

    dim3 grid(L_ / QTILE, H_, B_);   // (32, 8, 4)
    dim3 block(128);                  // 4 wave32 waves
    hipLaunchKernelGGL(MaskAttention_wmma_kernel, grid, block, 0, stream,
                       Q, K, V, Msk, O);
}
